// SpecialTransformer_45148696215768
// MI455X (gfx1250) — matrix operand order repaired and checked by reading, not compiled, not run on hardware
//
#include <hip/hip_runtime.h>

typedef __attribute__((ext_vector_type(16))) __bf16 v16bf;
typedef __attribute__((ext_vector_type(8)))  float  v8f;

union FragU { v16bf v; uint4 q[2]; };

constexpr int kB = 4, kS = 2048, kD = 1024, kH = 8, kDK = 128, kDFF = 2048;
constexpr int LDK = 40;    // GEMM LDS row pitch (ushorts): 32 data + 8 pad (80B rows, 16B aligned)
constexpr int LDA = 136;   // attention LDS row pitch (ushorts): 128 data + 8 pad (272B rows)

#if defined(__has_builtin)
#if __has_builtin(__builtin_amdgcn_global_load_async_to_lds_b128)
#define HAVE_ASYNC 1
#endif
#endif

// native RTNE fp32 -> bf16 (lets the backend emit v_cvt_*bf16_f32)
__device__ __forceinline__ unsigned short f2bf(float f) {
  __bf16 h = (__bf16)f;
  return __builtin_bit_cast(unsigned short, h);
}

__device__ __forceinline__ v8f v8f_zero() {
  v8f z;
  #pragma unroll
  for (int i = 0; i < 8; ++i) z[i] = 0.0f;
  return z;
}

// 16-byte global(bf16) -> LDS copy; async (ASYNCcnt) path when available.
__device__ __forceinline__ void cp16(const unsigned short* g, unsigned short* l) {
#ifdef HAVE_ASYNC
  typedef int v4i __attribute__((__vector_size__(4 * sizeof(int))));
  typedef __attribute__((address_space(1))) v4i* gp_t;
  typedef __attribute__((address_space(3))) v4i* lp_t;
  __builtin_amdgcn_global_load_async_to_lds_b128(
      (gp_t)(unsigned short*)g, (lp_t)l, 0, 0);
#else
  *(uint4*)l = *(const uint4*)g;
#endif
}
__device__ __forceinline__ void cp_wait() {
#ifdef HAVE_ASYNC
  asm volatile("s_wait_asynccnt 0x0" ::: "memory");
#endif
}

// A-fragment (16x32 bf16) from row-major LDS [row][k]
__device__ __forceinline__ v16bf frag_a(const unsigned short* base, int ld, int row, int k0, int hf) {
  FragU f;
  const unsigned short* p = base + row * ld + k0 + hf * 8;
  f.q[0] = *(const uint4*)(p);
  f.q[1] = *(const uint4*)(p + 16);
  return f.v;
}
// B-fragment (32x16 bf16) from [N][K]-major LDS [col][k]
__device__ __forceinline__ v16bf frag_b(const unsigned short* base, int ld, int col, int k0, int hf) {
  FragU f;
  const unsigned short* p = base + col * ld + k0 + hf * 8;
  f.q[0] = *(const uint4*)(p);
  f.q[1] = *(const uint4*)(p + 16);
  return f.v;
}

__device__ __forceinline__ v8f wmma_bf16(v16bf a, v16bf b, v8f c) {
  return __builtin_amdgcn_wmma_f32_16x16x32_bf16(false, a, false, b, (short)0, c, false, false);
}

// --- LDS tile fills -------------------------------------------------------
__device__ __forceinline__ void fill_a_tile(const float* A, int lda, int k0,
                                            unsigned short* As, int tid) {
  #pragma unroll
  for (int e = 0; e < 16; ++e) {         // 128x32 fp32 -> bf16, coalesced
    int f = tid + e * 256;
    int m = f >> 5, k = f & 31;
    As[m * LDK + k] = f2bf(A[(long)m * lda + k0 + k]);
  }
}
__device__ __forceinline__ void fill_a_tile(const unsigned short* A, int lda, int k0,
                                            unsigned short* As, int tid) {
  #pragma unroll
  for (int e = 0; e < 2; ++e) {          // 128x32 bf16: 512 16B chunks
    int f = tid + e * 256;
    int m = f >> 2, c = f & 3;
    cp16(A + (long)m * lda + k0 + c * 8, As + m * LDK + c * 8);
  }
}
__device__ __forceinline__ void fill_b_tile(const unsigned short* Bt, int ldb, int k0,
                                            unsigned short* Bs, int tid) {
  #pragma unroll
  for (int e = 0; e < 2; ++e) {          // pre-transposed [N][K] bf16 weights
    int f = tid + e * 256;
    int n = f >> 2, c = f & 3;
    cp16(Bt + (long)n * ldb + k0 + c * 8, Bs + n * LDK + c * 8);
  }
}

// ---------------------------------------------------------------------------
// 128x128 tile GEMM core: 8 waves in 4x2 grid, 32x64 per wave (2x4 WMMA accs).
// B is always pre-transposed bf16 [N][K]. Compile-time epilogue.
// ---------------------------------------------------------------------------
template <typename AT, bool BF16OUT, bool HAS_BIAS, bool RELU, bool TRANSOUT = false>
__device__ void gemm_tile_128(const AT* __restrict__ A, int lda,
                              const unsigned short* __restrict__ Bt, int ldb,
                              void* __restrict__ Cv, int ldc, int K,
                              const float* __restrict__ bias,
                              unsigned short* As, unsigned short* Bs) {
  const int tid  = threadIdx.x;
  const int lane = tid & 31;
  const int wave = tid >> 5;
  const int hf   = lane >> 4;
  const int l16  = lane & 15;
  const int wm   = (wave & 3) * 32;
  const int wn   = (wave >> 2) * 64;

  v8f acc[2][4];
  #pragma unroll
  for (int i = 0; i < 2; ++i)
    #pragma unroll
    for (int j = 0; j < 4; ++j) acc[i][j] = v8f_zero();

  for (int k0 = 0; k0 < K; k0 += 32) {
    __syncthreads();
    fill_a_tile(A, lda, k0, As, tid);
    fill_b_tile(Bt, ldb, k0, Bs, tid);
    cp_wait();
    __syncthreads();

    v16bf bf[4];
    #pragma unroll
    for (int j = 0; j < 4; ++j) bf[j] = frag_b(Bs, LDK, wn + j * 16 + l16, 0, hf);
    #pragma unroll
    for (int i = 0; i < 2; ++i) {
      v16bf af = frag_a(As, LDK, wm + i * 16 + l16, 0, hf);
      #pragma unroll
      for (int j = 0; j < 4; ++j) acc[i][j] = wmma_bf16(af, bf[j], acc[i][j]);
    }
  }

  float bcol[4];
  if constexpr (HAS_BIAS) {
    #pragma unroll
    for (int j = 0; j < 4; ++j) bcol[j] = bias[wn + j * 16 + l16];
  }
  #pragma unroll
  for (int i = 0; i < 2; ++i)
    #pragma unroll
    for (int j = 0; j < 4; ++j)
      #pragma unroll
      for (int v = 0; v < 8; ++v) {
        int row = wm + i * 16 + hf * 8 + v;
        int col = wn + j * 16 + l16;
        float x = acc[i][j][v];
        if constexpr (HAS_BIAS) x += bcol[j];
        if constexpr (RELU) x = fmaxf(x, 0.0f);
        long off = TRANSOUT ? (long)col * ldc + row : (long)row * ldc + col;
        if constexpr (BF16OUT) ((unsigned short*)Cv)[off] = f2bf(x);
        else                   ((float*)Cv)[off] = x;
      }
}

// ---------------------------------------------------------------------------
// Kernel 0: weight prep — transpose to [N][K] and convert to bf16.
// grid.x covers K*N elements, grid.y = batch (heads).
// ---------------------------------------------------------------------------
__global__ __launch_bounds__(256) void wprep_kernel(
    const float* __restrict__ in, unsigned short* __restrict__ out, int K, int N) {
  const long mat = (long)K * N;
  const float* ib = in + (long)blockIdx.y * mat;
  unsigned short* ob = out + (long)blockIdx.y * mat;
  long idx = (long)blockIdx.x * 256 + threadIdx.x;
  if (idx >= mat) return;
  int n = (int)(idx / K), k = (int)(idx % K);
  ob[idx] = f2bf(ib[(long)k * N + n]);   // out[n][k] = in[k][n]
}

// ---------------------------------------------------------------------------
// Kernel 1: per-head QKV projections -> bf16 heads; v written transposed [dv][s].
// grid = (S/128, 1, 3*B*H)
// ---------------------------------------------------------------------------
__global__ __launch_bounds__(256) void qkv_kernel(
    const float* __restrict__ query, const float* __restrict__ key,
    const float* __restrict__ value,
    const unsigned short* __restrict__ WqT, const unsigned short* __restrict__ WkT,
    const unsigned short* __restrict__ WvT,
    unsigned short* __restrict__ qbuf, unsigned short* __restrict__ kbuf,
    unsigned short* __restrict__ vbuf) {
  __shared__ __align__(16) unsigned short As[128 * LDK];
  __shared__ __align__(16) unsigned short Bs[128 * LDK];
  const int z = blockIdx.z;
  const int which = z >> 5;               // 0=q 1=k 2=v
  const int bh = z & 31;
  const int b = bh >> 3, h = bh & 7;
  const float* A = (which == 0 ? query : which == 1 ? key : value)
                   + (long)b * kS * kD + (long)blockIdx.x * 128 * kD;
  const unsigned short* Bt = (which == 0 ? WqT : which == 1 ? WkT : WvT)
                             + (long)h * kDK * kD;            // [dk][d], ldb = kD
  if (which == 2) {
    unsigned short* C = vbuf + (long)bh * kDK * kS + blockIdx.x * 128;   // [dv][s]
    gemm_tile_128<float, true, false, false, true>(A, kD, Bt, kD, C, kS, kD,
                                                   nullptr, As, Bs);
  } else {
    unsigned short* C = (which == 0 ? qbuf : kbuf)
                        + (long)bh * kS * kDK + (long)blockIdx.x * 128 * kDK;
    gemm_tile_128<float, true, false, false, false>(A, kD, Bt, kD, C, kDK, kD,
                                                    nullptr, As, Bs);
  }
}

// ---------------------------------------------------------------------------
// Kernel 2: fused flash attention, all-bf16 operands, async LDS fills.
// grid = (S/128, 1, B*H), 256 threads; wave owns 16 q-rows x 128 cols.
// ---------------------------------------------------------------------------
__global__ __launch_bounds__(256) void attn_kernel(
    const unsigned short* __restrict__ qbuf, const unsigned short* __restrict__ kbuf,
    const unsigned short* __restrict__ vbuf, const float* __restrict__ mask,
    unsigned short* __restrict__ obuf) {
  extern __shared__ __align__(16) unsigned short sm[];
  unsigned short* Qs = sm;                 // [128 s][128 dk]
  unsigned short* Ks = Qs + 128 * LDA;     // [128 t][128 dk]
  unsigned short* Vs = Ks + 128 * LDA;     // [128 dv][128 t]
  unsigned short* Ps = Vs + 128 * LDA;     // 8 waves x [16 s][128 t]
  float* Ms = (float*)(Ps + 128 * LDA);    // [128]

  const int tid = threadIdx.x;
  const int lane = tid & 31;
  const int wave = tid >> 5;
  const int hf = lane >> 4;
  const int l16 = lane & 15;

  const int bh = blockIdx.z;
  const int b = bh >> 3, h = bh & 7;
  const int s0 = blockIdx.x * 128;

  const unsigned short* qg = qbuf + (long)bh * kS * kDK;
  const unsigned short* kg = kbuf + (long)bh * kS * kDK;
  const unsigned short* vg = vbuf + (long)bh * kDK * kS;   // transposed [dv][s]

  #pragma unroll
  for (int e = 0; e < 8; ++e) {            // resident Q tile, 16B chunks
    int f = tid + e * 256;
    int r = f >> 4, c = f & 15;
    cp16(qg + (long)(s0 + r) * kDK + c * 8, Qs + r * LDA + c * 8);
  }

  unsigned short* Psw = Ps + wave * 16 * LDA;

  float mrow[8], lrow[8];
  v8f oa[8];
  #pragma unroll
  for (int v = 0; v < 8; ++v) { mrow[v] = -1e30f; lrow[v] = 0.0f; }
  #pragma unroll
  for (int j = 0; j < 8; ++j) oa[j] = v8f_zero();

  for (int t0 = 0; t0 < kS; t0 += 128) {
    __syncthreads();
    #pragma unroll
    for (int e = 0; e < 8; ++e) {          // K tile [t][dk]
      int f = tid + e * 256;
      int t = f >> 4, c = f & 15;
      cp16(kg + (long)(t0 + t) * kDK + c * 8, Ks + t * LDA + c * 8);
    }
    #pragma unroll
    for (int e = 0; e < 8; ++e) {          // V tile [dv][t] (already transposed)
      int f = tid + e * 256;
      int d = f >> 4, c = f & 15;
      cp16(vg + (long)d * kS + t0 + c * 8, Vs + d * LDA + c * 8);
    }
    if (tid < 128) Ms[tid] = mask[(long)b * kS + t0 + tid];
    cp_wait();
    __syncthreads();

    // S = Q * K^T : per wave 16x128, K-dim 128
    v8f sa[8];
    #pragma unroll
    for (int j = 0; j < 8; ++j) sa[j] = v8f_zero();
    #pragma unroll
    for (int kk = 0; kk < 128; kk += 32) {
      v16bf aq = frag_a(Qs, LDA, wave * 16 + l16, kk, hf);
      #pragma unroll
      for (int j = 0; j < 8; ++j) {
        v16bf bk = frag_b(Ks, LDA, j * 16 + l16, kk, hf);
        sa[j] = wmma_bf16(aq, bk, sa[j]);
      }
    }

    // mask: A = m*A + (1-m)*(-1e30)
    #pragma unroll
    for (int j = 0; j < 8; ++j) {
      float mk = Ms[j * 16 + l16];
      float off = (1.0f - mk) * -1e30f;
      #pragma unroll
      for (int v = 0; v < 8; ++v) sa[j][v] = mk * sa[j][v] + off;
    }

    // online softmax (rows = hf*8+v), 16-lane reductions
    #pragma unroll
    for (int v = 0; v < 8; ++v) {
      float mx = -1e30f;
      #pragma unroll
      for (int j = 0; j < 8; ++j) mx = fmaxf(mx, sa[j][v]);
      #pragma unroll
      for (int m = 1; m <= 8; m <<= 1) mx = fmaxf(mx, __shfl_xor(mx, m, 32));
      float mnew = fmaxf(mrow[v], mx);
      float corr = __expf(mrow[v] - mnew);
      mrow[v] = mnew;
      float rs = 0.0f;
      #pragma unroll
      for (int j = 0; j < 8; ++j) {
        float p = __expf(sa[j][v] - mnew);
        sa[j][v] = p;
        rs += p;
      }
      #pragma unroll
      for (int m = 1; m <= 8; m <<= 1) rs += __shfl_xor(rs, m, 32);
      lrow[v] = lrow[v] * corr + rs;
      #pragma unroll
      for (int j = 0; j < 8; ++j) oa[j][v] *= corr;
    }

    // P -> wave-private LDS (A-fragment layout)
    #pragma unroll
    for (int j = 0; j < 8; ++j)
      #pragma unroll
      for (int v = 0; v < 8; ++v)
        Psw[(hf * 8 + v) * LDA + j * 16 + l16] = f2bf(sa[j][v]);
    asm volatile("s_wait_dscnt 0" ::: "memory");

    // O += P * V
    #pragma unroll
    for (int kk = 0; kk < 128; kk += 32) {
      v16bf ap = frag_a(Psw, LDA, l16, kk, hf);
      #pragma unroll
      for (int j = 0; j < 8; ++j) {
        v16bf bv = frag_b(Vs, LDA, j * 16 + l16, kk, hf);
        oa[j] = wmma_bf16(ap, bv, oa[j]);
      }
    }
  }

  // normalize, write bf16 o[b][s][h*DV + dv]
  #pragma unroll
  for (int v = 0; v < 8; ++v) {
    float inv = lrow[v] > 0.0f ? 1.0f / lrow[v] : 0.0f;
    int srow = s0 + wave * 16 + hf * 8 + v;
    unsigned short* orow = obuf + ((long)b * kS + srow) * (kH * kDK) + h * kDK;
    #pragma unroll
    for (int j = 0; j < 8; ++j) orow[j * 16 + l16] = f2bf(oa[j][v] * inv);
  }
}

// ---------------------------------------------------------------------------
// Kernel 3: generic GEMM wrapper. grid = (M/128, N/128)
// ---------------------------------------------------------------------------
template <typename AT, bool BF16OUT, bool HAS_BIAS, bool RELU>
__global__ __launch_bounds__(256) void gemm_kernel(
    const AT* __restrict__ A, const unsigned short* __restrict__ Bt,
    void* __restrict__ C, int lda, int ldb, int ldc, int K,
    const float* __restrict__ bias) {
  __shared__ __align__(16) unsigned short As[128 * LDK];
  __shared__ __align__(16) unsigned short Bs[128 * LDK];
  const long mBase = (long)blockIdx.x * 128;
  const long nBase = (long)blockIdx.y * 128;
  void* Ct;
  if constexpr (BF16OUT) Ct = (void*)((unsigned short*)C + mBase * ldc + nBase);
  else                   Ct = (void*)((float*)C + mBase * ldc + nBase);
  gemm_tile_128<AT, BF16OUT, HAS_BIAS, RELU>(
      A + mBase * lda, lda, Bt + nBase * ldb, ldb, Ct, ldc, K,
      HAS_BIAS ? bias + nBase : nullptr, As, Bs);
}

// ---------------------------------------------------------------------------
// Kernel 4: fused residual + LayerNorm; optional bf16 shadow copy for GEMM A.
// ---------------------------------------------------------------------------
__global__ __launch_bounds__(256) void ln_kernel(
    const float* __restrict__ X, const float* __restrict__ Y,
    float* __restrict__ Outf, unsigned short* __restrict__ Outb,
    const float* __restrict__ gamma, const float* __restrict__ beta, int gi) {
  __shared__ float red[2][8];
  const int tid = threadIdx.x;
  const long base = (long)blockIdx.x * kD;
  float x[4];
  float s = 0.0f, ss = 0.0f;
  #pragma unroll
  for (int k = 0; k < 4; ++k) {
    int e = tid + k * 256;
    float v = X[base + e] + Y[base + e];
    x[k] = v; s += v; ss += v * v;
  }
  #pragma unroll
  for (int m = 1; m <= 16; m <<= 1) {
    s  += __shfl_xor(s,  m, 32);
    ss += __shfl_xor(ss, m, 32);
  }
  if ((tid & 31) == 0) { red[0][tid >> 5] = s; red[1][tid >> 5] = ss; }
  __syncthreads();
  float ts = 0.0f, tss = 0.0f;
  #pragma unroll
  for (int w = 0; w < 8; ++w) { ts += red[0][w]; tss += red[1][w]; }
  float mean = ts * (1.0f / kD);
  float var = tss * (1.0f / kD) - mean * mean;
  float inv = rsqrtf(var + 1e-14f);
  float g = gamma[gi], bb = beta[gi];
  #pragma unroll
  for (int k = 0; k < 4; ++k) {
    int e = tid + k * 256;
    float r = (x[k] - mean) * inv * g + bb;
    Outf[base + e] = r;
    if (Outb) Outb[base + e] = f2bf(r);
  }
}

// ---------------------------------------------------------------------------
extern "C" void kernel_launch(void* const* d_in, const int* in_sizes, int n_in,
                              void* d_out, int out_size, void* d_ws, size_t ws_size,
                              hipStream_t stream) {
  (void)in_sizes; (void)n_in; (void)out_size; (void)ws_size;
  const float* query = (const float*)d_in[0];
  const float* key   = (const float*)d_in[1];
  const float* value = (const float*)d_in[2];
  const float* mask  = (const float*)d_in[3];
  const float* Wq = (const float*)d_in[4];
  const float* Wk = (const float*)d_in[5];
  const float* Wv = (const float*)d_in[6];
  const float* Wo = (const float*)d_in[7];
  const float* W1 = (const float*)d_in[8];
  const float* b1 = (const float*)d_in[9];
  const float* W2 = (const float*)d_in[10];
  const float* b2 = (const float*)d_in[11];
  const float* gamma = (const float*)d_in[12];
  const float* beta  = (const float*)d_in[13];
  float* out = (float*)d_out;

  // ---- workspace layout (256B aligned chunks) ----
  char* p = (char*)d_ws;
  auto alloc = [&](size_t bytes) {
    char* r = p;
    p += (bytes + 255) & ~(size_t)255;
    return r;
  };
  const long NH = (long)kB * kH * kS * kDK;         // 8,388,608 elements
  unsigned short* wqT = (unsigned short*)alloc((size_t)kH * kDK * kD * 2);
  unsigned short* wkT = (unsigned short*)alloc((size_t)kH * kDK * kD * 2);
  unsigned short* wvT = (unsigned short*)alloc((size_t)kH * kDK * kD * 2);
  unsigned short* woT = (unsigned short*)alloc((size_t)kD * kD * 2);
  unsigned short* w1T = (unsigned short*)alloc((size_t)kDFF * kD * 2);
  unsigned short* w2T = (unsigned short*)alloc((size_t)kD * kDFF * 2);
  unsigned short* qbuf = (unsigned short*)alloc((size_t)NH * 2);
  unsigned short* kbuf = (unsigned short*)alloc((size_t)NH * 2);
  unsigned short* vbuf = (unsigned short*)alloc((size_t)NH * 2);
  unsigned short* obuf = (unsigned short*)alloc((size_t)NH * 2);
  unsigned short* x1b  = (unsigned short*)alloc((size_t)NH * 2);
  unsigned short* h1   = (unsigned short*)alloc((size_t)kB * kS * kDFF * 2);
  float* pfbuf = (float*)alloc((size_t)NH * 4);
  float* x1f   = (float*)alloc((size_t)NH * 4);

  dim3 blk(256);

  // 0) weight prep: transpose + bf16
  wprep_kernel<<<dim3(512, kH), blk, 0, stream>>>(Wq, wqT, kD, kDK);
  wprep_kernel<<<dim3(512, kH), blk, 0, stream>>>(Wk, wkT, kD, kDK);
  wprep_kernel<<<dim3(512, kH), blk, 0, stream>>>(Wv, wvT, kD, kDK);
  wprep_kernel<<<dim3(4096, 1), blk, 0, stream>>>(Wo, woT, kD, kD);
  wprep_kernel<<<dim3(8192, 1), blk, 0, stream>>>(W1, w1T, kD, kDFF);
  wprep_kernel<<<dim3(8192, 1), blk, 0, stream>>>(W2, w2T, kDFF, kD);

  // 1) q/k/v head projections (v transposed)
  qkv_kernel<<<dim3(kS / 128, 1, 3 * kB * kH), blk, 0, stream>>>(
      query, key, value, wqT, wkT, wvT, qbuf, kbuf, vbuf);

  // 2) fused attention
  size_t smem = (size_t)(4 * 128 * LDA) * sizeof(unsigned short) + 128 * sizeof(float);
  attn_kernel<<<dim3(kS / 128, 1, kB * kH), blk, smem, stream>>>(
      qbuf, kbuf, vbuf, mask, obuf);

  // 3) proj = o @ Wo (bf16 A, fp32 out)
  gemm_kernel<unsigned short, false, false, false>
      <<<dim3((kB * kS) / 128, kD / 128), blk, 0, stream>>>(
          obuf, woT, pfbuf, kH * kDK, kD, kD, kD, nullptr);

  // 4) x1 = LN(value + proj), fp32 + bf16 shadow
  ln_kernel<<<dim3(kB * kS), blk, 0, stream>>>(value, pfbuf, x1f, x1b, gamma, beta, 0);

  // 5) h1 = relu(x1 @ W1 + b1), bf16 out
  gemm_kernel<unsigned short, true, true, true>
      <<<dim3((kB * kS) / 128, kDFF / 128), blk, 0, stream>>>(
          x1b, w1T, h1, kD, kD, kDFF, kD, b1);

  // 6) ffn = h1 @ W2 + b2, fp32 out
  gemm_kernel<unsigned short, false, true, false>
      <<<dim3((kB * kS) / 128, kD / 128), blk, 0, stream>>>(
          h1, w2T, pfbuf, kDFF, kDFF, kD, kDFF, b2);

  // 7) out = LN(x1 + ffn)
  ln_kernel<<<dim3(kB * kS), blk, 0, stream>>>(x1f, pfbuf, out, nullptr, gamma, beta, 1);
}
